// Attention_90735479095384
// MI455X (gfx1250) — compile-verified, matrix operand order checked by reading, not run on hardware
//
#include <hip/hip_runtime.h>
#include <hip/hip_bf16.h>
#include <math.h>

// ---------------------------------------------------------------------------
// Problem constants (from reference)
// ---------------------------------------------------------------------------
#define B_      2
#define SEQ_    2048
#define DIM_    768
#define HEADS_  8
#define DKEY_   64
#define DINNER_ 512          // HEADS_*DKEY_
#define RELF_   192          // NUM_REL_POS_FEATURES
#define RELN_   4095         // 2*SEQ-1
#define RELPAD_ 4096         // padded rows (row 4095 = zeros)
#define MTOT_   (B_*SEQ_)    // 4096

// ---------------------------------------------------------------------------
// WMMA types / helpers (CDNA5 gfx1250, wave32)
// ---------------------------------------------------------------------------
typedef __attribute__((ext_vector_type(16))) __bf16 v16bf;
typedef __attribute__((ext_vector_type(8)))  __bf16 v8bf;
typedef __attribute__((ext_vector_type(8)))  float  v8f;

__device__ __forceinline__ v8f wmma_bf16(v16bf a, v16bf b, v8f c) {
  // v_wmma_f32_16x16x32_bf16  (neg_a, A, neg_b, B, c_mod, C, reuse_a, reuse_b)
  return __builtin_amdgcn_wmma_f32_16x16x32_bf16(false, a, false, b, (short)0, c, false, false);
}

// Load a 16x32 bf16 fragment (A layout; B layout is the mirror with rows of a
// pre-transposed [N x K] operand). Lane L (L<16): row r0+L, K = kk..kk+7 and
// kk+16..kk+23.  Lane L>=16: row r0+(L-16), K = kk+8.. and kk+24.. .
// Two b128 loads per lane (global_load_b128 or ds_load_b128 depending on AS).
__device__ __forceinline__ v16bf load_frag(const __bf16* __restrict__ base,
                                           int stride, int r0, int kk, int lane) {
  int ln = lane & 15, hi = lane >> 4;
  const __bf16* p = base + (size_t)(r0 + ln) * (size_t)stride + kk + hi * 8;
  v8bf c0 = *(const v8bf*)p;
  v8bf c1 = *(const v8bf*)(p + 16);
  v16bf r;
#pragma unroll
  for (int i = 0; i < 8; ++i) { r[i] = c0[i]; r[i + 8] = c1[i]; }
  return r;
}

__device__ __forceinline__ float redmax16(float v) {
#pragma unroll
  for (int m = 1; m <= 8; m <<= 1) v = fmaxf(v, __shfl_xor(v, m, 32));
  return v;
}
__device__ __forceinline__ float redsum16(float v) {
#pragma unroll
  for (int m = 1; m <= 8; m <<= 1) v += __shfl_xor(v, m, 32);
  return v;
}

// ---------------------------------------------------------------------------
// CDNA5 async Global -> LDS copy (ASYNCcnt path).  Per-lane 16B transfer:
//   LDS[vdst_lane] = MEM[vaddr_lane .. +15]
// Flat shared pointers carry the LDS offset in their low 32 bits.
// ---------------------------------------------------------------------------
__device__ __forceinline__ void async_copy16(void* lds_ptr, const void* gptr) {
  unsigned lds = (unsigned)(uintptr_t)lds_ptr;   // LDS byte address
  asm volatile("global_load_async_to_lds_b128 %0, %1, off"
               :: "v"(lds), "v"(gptr)
               : "memory");
}
__device__ __forceinline__ void wait_async0() {
  asm volatile("s_wait_asynccnt 0" ::: "memory");
}

// ---------------------------------------------------------------------------
// Elementwise / layout kernels
// ---------------------------------------------------------------------------
__global__ void cvt_f32_bf16_kernel(const float* __restrict__ s, __bf16* __restrict__ d, int n) {
  int i = blockIdx.x * blockDim.x + threadIdx.x;
  if (i < n) d[i] = (__bf16)s[i];
}

// wt[n*K + k] = w[k*N + n]  (transpose + convert so WMMA B-operands load contiguously)
__global__ void wtrans_kernel(const float* __restrict__ w, __bf16* __restrict__ wt, int K, int N) {
  int i = blockIdx.x * blockDim.x + threadIdx.x;
  if (i >= K * N) return;
  int n = i % N, k = i / N;
  wt[(size_t)n * K + k] = (__bf16)w[(size_t)k * N + n];
}

// Positional embedding rows [RELPAD_][RELF_] (bf16), row 4095 zeroed.
__global__ void pos_embed_kernel(__bf16* __restrict__ pos) {
  int r = blockIdx.x * blockDim.x + threadIdx.x;
  if (r >= RELPAD_) return;
  __bf16* row = pos + (size_t)r * RELF_;
  if (r >= RELN_) {
    for (int i = 0; i < RELF_; ++i) row[i] = (__bf16)0.0f;
    return;
  }
  float d    = (float)(r - (SEQ_ - 1));
  float absd = fabsf(d);
  float sgn  = (d > 0.f) ? 1.f : (d < 0.f ? -1.f : 0.f);
  float feat[96];
  // exponential basis: 2^(-|d| / 2^linspace(3, log2(SEQ), 32))
#pragma unroll 4
  for (int i = 0; i < 32; ++i) {
    float e = 3.f + 8.f * (float)i / 31.f;     // log2(2048)=11 -> span 8
    feat[i] = exp2f(-absd * exp2f(-e));
  }
  // central mask basis
#pragma unroll 4
  for (int i = 0; i < 32; ++i) {
    float cw = exp2f((float)(i + 1)) - 1.f;
    feat[32 + i] = (cw > absd) ? 1.f : 0.f;
  }
  // gamma pdf basis (stddev=32, mean_i = 64 + 64 i)
  float gmax = 0.f;
#pragma unroll 4
  for (int i = 0; i < 32; ++i) {
    float mean = 64.f + 64.f * (float)i;
    float ms   = mean / 32.f;
    float conc = ms * ms;
    float rate = mean / 1024.f;
    float lu    = (conc - 1.f) * logf(absd) - rate * absd;   // -> -inf at d=0
    float lnorm = lgammaf(conc) - conc * logf(rate);
    float p = expf(lu - lnorm) + 1e-8f;
    feat[64 + i] = p;
    gmax = fmaxf(gmax, p);
  }
  float inv = 1.f / gmax;
#pragma unroll 4
  for (int i = 0; i < 32; ++i) feat[64 + i] *= inv;
#pragma unroll 4
  for (int i = 0; i < 96; ++i) {
    row[i]      = (__bf16)feat[i];
    row[96 + i] = (__bf16)(sgn * feat[i]);
  }
}

// qc = bf16(q*scale + rel_content_bias[h,d]); qp = bf16(q*scale + rel_pos_bias[h,d])
__global__ void qbias_kernel(const float* __restrict__ q,
                             const float* __restrict__ rcb, const float* __restrict__ rpb,
                             __bf16* __restrict__ qc, __bf16* __restrict__ qp) {
  int i = blockIdx.x * blockDim.x + threadIdx.x;
  if (i >= MTOT_ * DINNER_) return;
  int c = i & (DINNER_ - 1);
  float v = q[i] * 0.125f;                      // DIM_KEY^-0.5
  qc[i] = (__bf16)(v + rcb[c]);
  qp[i] = (__bf16)(v + rpb[c]);
}

// vt[b][h][d][n] = v[(b*SEQ+n)*512 + h*64 + d]
__global__ void vtrans_kernel(const __bf16* __restrict__ v, __bf16* __restrict__ vt) {
  int i = blockIdx.x * blockDim.x + threadIdx.x;
  if (i >= B_ * HEADS_ * DKEY_ * SEQ_) return;
  int n = i & (SEQ_ - 1);
  int d = (i >> 11) & 63;
  int h = (i >> 17) & 7;
  int b = i >> 20;
  vt[i] = v[((size_t)(b * SEQ_ + n)) * DINNER_ + h * DKEY_ + d];
}

// ---------------------------------------------------------------------------
// Generic bf16 WMMA GEMM: C[M x N] = A[M x K] * Bt[N x K]^T  (Bt pre-transposed)
// One wave computes a 16x64 tile.  EPI: 0 = f32 out, 1 = bf16 out, 2 = f32 + bias
// ---------------------------------------------------------------------------
template <int EPI>
__global__ __launch_bounds__(256) void gemm_bf16_kernel(
    const __bf16* __restrict__ A, const __bf16* __restrict__ Bt,
    float* __restrict__ outF, __bf16* __restrict__ outB,
    const float* __restrict__ bias, int M, int N, int K) {
  int lane = threadIdx.x & 31, wave = threadIdx.x >> 5;
  int g = blockIdx.x * 8 + wave;
  int ntiles = N >> 6;
  if (g >= (M >> 4) * ntiles) return;
  int m0 = (g / ntiles) << 4;
  int n0 = (g % ntiles) << 6;
  v8f acc[4] = {v8f{}, v8f{}, v8f{}, v8f{}};
  for (int kk = 0; kk < K; kk += 32) {
    if (kk + 32 < K)
      __builtin_prefetch(A + (size_t)(m0 + (lane & 15)) * K + kk + 32, 0, 1);
    v16bf a = load_frag(A, K, m0, kk, lane);
#pragma unroll
    for (int t = 0; t < 4; ++t) {
      v16bf b = load_frag(Bt, K, n0 + t * 16, kk, lane);
      acc[t] = wmma_bf16(a, b, acc[t]);
    }
  }
  int hi = lane >> 4, ln = lane & 15;
#pragma unroll
  for (int t = 0; t < 4; ++t) {
#pragma unroll
    for (int r = 0; r < 8; ++r) {
      int row = m0 + r + 8 * hi, col = n0 + t * 16 + ln;
      float v = acc[t][r];
      if (EPI == 2) v += bias[col];
      if (EPI == 1) outB[(size_t)row * N + col] = (__bf16)v;
      else          outF[(size_t)row * N + col] = v;
    }
  }
}

// ---------------------------------------------------------------------------
// Fused relative-position flash attention.
// All 8 waves of a block share one (b,h); K / V^T tiles for each 32-key step
// are staged once per block into LDS with async Global->LDS copies (double-
// buffered, ASYNCcnt), shared by all waves.  Each wave owns 16 query rows.
// logits[i,j] = Qc[i]·K[j] + Qp[i]·RelK[j-i+N-1]   (Transformer-XL shift)
// ---------------------------------------------------------------------------
__global__ __launch_bounds__(256) void attn_kernel(
    const __bf16* __restrict__ qc, const __bf16* __restrict__ qp,
    const __bf16* __restrict__ kmat, const __bf16* __restrict__ vt,
    const __bf16* __restrict__ relk, __bf16* __restrict__ out) {
  __shared__ float  sStrip[8][16][32];   // per-wave rel strip (f32)
  __shared__ __bf16 sP[8][16][32];       // per-wave P tile (bf16)
  __shared__ __bf16 sK[2][32][64];       // block-shared K tile (keys x dims)
  __shared__ __bf16 sV[2][64][32];       // block-shared V^T tile (dims x keys)

  int tid  = threadIdx.x;
  int lane = tid & 31, wave = tid >> 5;
  int g  = blockIdx.x * 8 + wave;        // 2048 wave-tasks total
  int ib = g & 127;
  int h  = (g >> 7) & 7;                 // identical for all waves in block
  int b  = g >> 10;
  int i0 = ib << 4;
  int hi = lane >> 4, ln = lane & 15;

  const __bf16* qcb = qc   + (size_t)b * SEQ_ * DINNER_ + h * DKEY_;
  const __bf16* qpb = qp   + (size_t)b * SEQ_ * DINNER_ + h * DKEY_;
  const __bf16* kb  = kmat + (size_t)b * SEQ_ * DINNER_ + h * DKEY_;
  const __bf16* vtb = vt   + ((size_t)(b * HEADS_ + h)) * DKEY_ * SEQ_;
  const __bf16* rb  = relk + h * DKEY_;

  // cooperative async stage of one 32-key tile: 256 threads x 16B per array
  int krow = tid >> 3, kch = (tid & 7) * 8;   // 32 rows x 128B
  int vrow = tid >> 2, vch = (tid & 3) * 8;   // 64 rows x  64B
  auto stage = [&](int j0, int buf) {
    async_copy16(&sK[buf][krow][kch], kb + (size_t)(j0 + krow) * DINNER_ + kch);
    async_copy16(&sV[buf][vrow][vch], vtb + (size_t)vrow * SEQ_ + j0 + vch);
  };

  // Q fragments are loop-invariant: load once.
  v16bf aQc0 = load_frag(qcb, DINNER_, i0, 0,  lane);
  v16bf aQc1 = load_frag(qcb, DINNER_, i0, 32, lane);
  v16bf aQp0 = load_frag(qpb, DINNER_, i0, 0,  lane);
  v16bf aQp1 = load_frag(qpb, DINNER_, i0, 32, lane);

  float m[8], l[8];
  v8f acc[4] = {v8f{}, v8f{}, v8f{}, v8f{}};
#pragma unroll
  for (int r = 0; r < 8; ++r) { m[r] = -1e30f; l[r] = 0.f; }

  stage(0, 0);                                 // prologue: tile 0 -> buf 0

  for (int j0 = 0; j0 < SEQ_; j0 += 32) {
    int cur = (j0 >> 5) & 1;
    wait_async0();                             // own tile-j0 copies done
    __syncthreads();                           // -> done block-wide; prev buf free
    if (j0 + 32 < SEQ_) stage(j0 + 32, cur ^ 1);   // overlap next copy w/ compute

    // ---- content logits: two 16x16 key tiles from LDS, K-dim 64 = 2 steps
    v8f S0 = v8f{}, S1 = v8f{};
    {
      const __bf16* kt = &sK[cur][0][0];
      v16bf bk;
      bk = load_frag(kt, 64, 0,  0,  lane); S0 = wmma_bf16(aQc0, bk, S0);
      bk = load_frag(kt, 64, 0,  32, lane); S0 = wmma_bf16(aQc1, bk, S0);
      bk = load_frag(kt, 64, 16, 0,  lane); S1 = wmma_bf16(aQc0, bk, S1);
      bk = load_frag(kt, 64, 16, 32, lane); S1 = wmma_bf16(aQc1, bk, S1);
    }

    // ---- relative logits: 32-wide strip per 16-key subtile, shift via LDS
#pragma unroll
    for (int sub = 0; sub < 2; ++sub) {
      int base = (j0 + sub * 16) - i0 + (SEQ_ - 16);   // strip col c = 15 + dj - di
      if (sub == 0 && j0 + 32 < SEQ_)                  // hint next band into cache
        __builtin_prefetch(rb + (size_t)(base + 32 + ln) * DINNER_, 0, 1);
      v8f T0 = v8f{}, T1 = v8f{};
      v16bf br;
      br = load_frag(rb, DINNER_, base,      0,  lane); T0 = wmma_bf16(aQp0, br, T0);
      br = load_frag(rb, DINNER_, base,      32, lane); T0 = wmma_bf16(aQp1, br, T0);
      br = load_frag(rb, DINNER_, base + 16, 0,  lane); T1 = wmma_bf16(aQp0, br, T1);
      br = load_frag(rb, DINNER_, base + 16, 32, lane); T1 = wmma_bf16(aQp1, br, T1);
#pragma unroll
      for (int r = 0; r < 8; ++r) {
        sStrip[wave][r + 8 * hi][ln]      = T0[r];
        sStrip[wave][r + 8 * hi][16 + ln] = T1[r];
      }
      // same-wave LDS ops are in-order (DScnt): safe to read back immediately
#pragma unroll
      for (int r = 0; r < 8; ++r) {
        int row = r + 8 * hi;
        float sh = sStrip[wave][row][15 + ln - row];
        if (sub == 0) S0[r] += sh; else S1[r] += sh;
      }
    }

    // ---- online softmax over this 32-key tile
#pragma unroll
    for (int r = 0; r < 8; ++r) {
      int row = r + 8 * hi;
      float rmax = redmax16(fmaxf(S0[r], S1[r]));
      float nm   = fmaxf(m[r], rmax);
      float corr = __expf(m[r] - nm);
      m[r] = nm;
      float p0 = __expf(S0[r] - nm);
      float p1 = __expf(S1[r] - nm);
      sP[wave][row][ln]      = (__bf16)p0;
      sP[wave][row][16 + ln] = (__bf16)p1;
      l[r] = l[r] * corr + redsum16(p0 + p1);
#pragma unroll
      for (int t = 0; t < 4; ++t) acc[t][r] *= corr;
    }

    // ---- P (C-layout) -> A-fragment via LDS, then P·V from staged V^T tile
    v16bf aP;
    {
      const __bf16* pp = &sP[wave][ln][hi * 8];
      v8bf c0 = *(const v8bf*)pp;
      v8bf c1 = *(const v8bf*)(pp + 16);
#pragma unroll
      for (int i = 0; i < 8; ++i) { aP[i] = c0[i]; aP[i + 8] = c1[i]; }
    }
    {
      const __bf16* vtt = &sV[cur][0][0];
#pragma unroll
      for (int t = 0; t < 4; ++t) {
        v16bf bv = load_frag(vtt, 32, t * 16, 0, lane);
        acc[t] = wmma_bf16(aP, bv, acc[t]);
      }
    }
  }

  // ---- epilogue: normalize, write bf16 [b*SEQ + i][h*64 + d]
#pragma unroll
  for (int r = 0; r < 8; ++r) {
    float inv = 1.0f / l[r];
    int row = i0 + r + 8 * hi;
#pragma unroll
    for (int t = 0; t < 4; ++t)
      out[(size_t)(b * SEQ_ + row) * DINNER_ + h * DKEY_ + t * 16 + ln] =
          (__bf16)(acc[t][r] * inv);
  }
}

// ---------------------------------------------------------------------------
// Host orchestration
// ---------------------------------------------------------------------------
extern "C" void kernel_launch(void* const* d_in, const int* in_sizes, int n_in,
                              void* d_out, int out_size, void* d_ws, size_t ws_size,
                              hipStream_t stream) {
  const float* x    = (const float*)d_in[0];
  const float* Wq   = (const float*)d_in[1];
  const float* Wk   = (const float*)d_in[2];
  const float* Wv   = (const float*)d_in[3];
  const float* Wrel = (const float*)d_in[4];
  const float* rcb  = (const float*)d_in[5];
  const float* rpb  = (const float*)d_in[6];
  const float* Wo   = (const float*)d_in[7];
  const float* bo   = (const float*)d_in[8];
  float* outp = (float*)d_out;

  char* ws = (char*)d_ws;
  size_t off = 0;
  auto alloc = [&](size_t bytes) -> void* {
    void* p = ws + off;
    off += (bytes + 255) & ~(size_t)255;
    return p;
  };
  __bf16* xb    = (__bf16*)alloc((size_t)MTOT_ * DIM_ * 2);
  __bf16* WqT   = (__bf16*)alloc((size_t)DINNER_ * DIM_ * 2);
  __bf16* WkT   = (__bf16*)alloc((size_t)DINNER_ * DIM_ * 2);
  __bf16* WvT   = (__bf16*)alloc((size_t)DINNER_ * DIM_ * 2);
  __bf16* WrelT = (__bf16*)alloc((size_t)DINNER_ * RELF_ * 2);
  __bf16* WoT   = (__bf16*)alloc((size_t)DIM_ * DINNER_ * 2);
  __bf16* posb  = (__bf16*)alloc((size_t)RELPAD_ * RELF_ * 2);
  float*  qf    = (float*)alloc((size_t)MTOT_ * DINNER_ * 4);
  __bf16* kbm   = (__bf16*)alloc((size_t)MTOT_ * DINNER_ * 2);
  __bf16* vbm   = (__bf16*)alloc((size_t)MTOT_ * DINNER_ * 2);
  __bf16* relkb = (__bf16*)alloc((size_t)RELPAD_ * DINNER_ * 2);
  __bf16* qcb   = (__bf16*)alloc((size_t)MTOT_ * DINNER_ * 2);
  __bf16* qpb   = (__bf16*)alloc((size_t)MTOT_ * DINNER_ * 2);
  __bf16* vtb   = (__bf16*)alloc((size_t)B_ * HEADS_ * DKEY_ * SEQ_ * 2);
  __bf16* attO  = (__bf16*)alloc((size_t)MTOT_ * DINNER_ * 2);

  const int T = 256;
  // 1. conversions / transposes
  cvt_f32_bf16_kernel<<<(MTOT_ * DIM_ + T - 1) / T, T, 0, stream>>>(x, xb, MTOT_ * DIM_);
  wtrans_kernel<<<(DIM_ * DINNER_ + T - 1) / T, T, 0, stream>>>(Wq, WqT, DIM_, DINNER_);
  wtrans_kernel<<<(DIM_ * DINNER_ + T - 1) / T, T, 0, stream>>>(Wk, WkT, DIM_, DINNER_);
  wtrans_kernel<<<(DIM_ * DINNER_ + T - 1) / T, T, 0, stream>>>(Wv, WvT, DIM_, DINNER_);
  wtrans_kernel<<<(RELF_ * DINNER_ + T - 1) / T, T, 0, stream>>>(Wrel, WrelT, RELF_, DINNER_);
  wtrans_kernel<<<(DINNER_ * DIM_ + T - 1) / T, T, 0, stream>>>(Wo, WoT, DINNER_, DIM_);
  pos_embed_kernel<<<(RELPAD_ + T - 1) / T, T, 0, stream>>>(posb);

  // 2. projection GEMMs (WMMA bf16): waves = (M/16)*(N/64), 8 waves/block
  int gqkv = ((MTOT_ >> 4) * (DINNER_ >> 6)) / 8;                 // 256
  gemm_bf16_kernel<0><<<gqkv, T, 0, stream>>>(xb, WqT, qf, nullptr, nullptr,
                                              MTOT_, DINNER_, DIM_);
  gemm_bf16_kernel<1><<<gqkv, T, 0, stream>>>(xb, WkT, nullptr, kbm, nullptr,
                                              MTOT_, DINNER_, DIM_);
  gemm_bf16_kernel<1><<<gqkv, T, 0, stream>>>(xb, WvT, nullptr, vbm, nullptr,
                                              MTOT_, DINNER_, DIM_);
  int grel = ((RELPAD_ >> 4) * (DINNER_ >> 6)) / 8;               // 256
  gemm_bf16_kernel<1><<<grel, T, 0, stream>>>(posb, WrelT, nullptr, relkb, nullptr,
                                              RELPAD_, DINNER_, RELF_);

  // 3. bias/scale + V transpose
  qbias_kernel<<<(MTOT_ * DINNER_ + T - 1) / T, T, 0, stream>>>(qf, rcb, rpb, qcb, qpb);
  vtrans_kernel<<<(B_ * HEADS_ * DKEY_ * SEQ_ + T - 1) / T, T, 0, stream>>>(vbm, vtb);

  // 4. fused relative-position flash attention (2048 wave-tasks, 8/block)
  attn_kernel<<<2048 / 8, T, 0, stream>>>(qcb, qpb, kbm, vtb, relkb, attO);

  // 5. output projection + bias -> d_out (f32)
  int gout = ((MTOT_ >> 4) * (DIM_ >> 6)) / 8;                    // 384
  gemm_bf16_kernel<2><<<gout, T, 0, stream>>>(attO, WoT, outp, nullptr, bo,
                                              MTOT_, DIM_, DINNER_);
}
